// MultiHeadSpatialAttention_47579647705483
// MI455X (gfx1250) — compile-verified, matrix operand order checked by reading, not run on hardware
//
#include <hip/hip_runtime.h>

typedef __bf16 bf16;
typedef bf16  bf16x16 __attribute__((ext_vector_type(16)));
typedef bf16  bf16x8  __attribute__((ext_vector_type(8)));
typedef float f32x8   __attribute__((ext_vector_type(8)));
typedef float f32x4   __attribute__((ext_vector_type(4)));

union FragAB { bf16x16 v; bf16x8 h[2]; };

#define BATCH 512
#define DIM   4096   // D = H*F
#define NHEAD 64
#define FDIM  64

// ---------------------------------------------------------------------------
// Kernel 1: fused QKV GEMM.  y = x @ W^T  for W in {Wq, Wk, Wv}.
// x:(512,4096) f32 row-major, W:(4096,4096) f32 row-major => both operands
// are contraction-contiguous.  Output bf16: q,k as (B,D); v transposed (B,F,H).
// Tile: 128(M) x 128(N), K-step 32, double-buffered LDS, 8 waves (2x4).
// ---------------------------------------------------------------------------
__launch_bounds__(256)
__global__ void qkv_gemm_wmma(const float* __restrict__ x,
                              const float* __restrict__ Wq,
                              const float* __restrict__ Wk,
                              const float* __restrict__ Wv,
                              bf16* __restrict__ qb,
                              bf16* __restrict__ kb,
                              bf16* __restrict__ vtb)
{
    constexpr int LDT = 40;                 // padded bf16 row stride (80B, 16B-aligned)
    __shared__ bf16 As[2][128 * LDT];
    __shared__ bf16 Bs[2][128 * LDT];

    const int tid  = threadIdx.x;
    const int row0 = blockIdx.x * 128;             // M block (batch rows)
    const int wsel = blockIdx.y >> 5;              // 0=Q, 1=K, 2=V
    const int col0 = (blockIdx.y & 31) * 128;      // N block

    const float* W = (wsel == 0) ? Wq : (wsel == 1) ? Wk : Wv;

    // global->reg staging: each thread owns 16 f32 of A and 16 f32 of B
    const int ldRow = tid >> 1;                    // 0..127
    const int ldCol = (tid & 1) * 16;              // 0 or 16
    f32x4 ra[4], rb[4];

    auto gload = [&](int k0) {
        const float* ap = x + (size_t)(row0 + ldRow) * DIM + k0 + ldCol;
        const float* bp = W + (size_t)(col0 + ldRow) * DIM + k0 + ldCol;
#pragma unroll
        for (int j = 0; j < 4; ++j) {
            ra[j] = *(const f32x4*)(ap + 4 * j);
            rb[j] = *(const f32x4*)(bp + 4 * j);
        }
    };
    auto sstore = [&](int buf) {
        bf16* arow = &As[buf][ldRow * LDT + ldCol];
        bf16* brow = &Bs[buf][ldRow * LDT + ldCol];
#pragma unroll
        for (int j = 0; j < 4; ++j)
#pragma unroll
            for (int e = 0; e < 4; ++e) {
                arow[4 * j + e] = (bf16)ra[j][e];
                brow[4 * j + e] = (bf16)rb[j][e];
            }
    };

    const int lane  = tid & 31;
    const int wv    = tid >> 5;       // 8 waves
    const int mbase = (wv & 1) * 64;  // wave M offset in tile
    const int nbase = (wv >> 1) * 32; // wave N offset in tile
    const int l     = lane & 15;
    const int hi    = lane >> 4;      // half-wave select (ISA K grouping)

    f32x8 acc[4][2] = {};

    gload(0);
    sstore(0);
    __syncthreads();

    const int NK = DIM / 32;
    for (int kt = 0; kt < NK; ++kt) {
        const int buf = kt & 1;
        if (kt + 1 < NK) gload((kt + 1) * 32);
        if (kt + 2 < NK) {                              // -> global_prefetch_b8
            __builtin_prefetch(x + (size_t)(row0 + ldRow) * DIM + (kt + 2) * 32 + ldCol, 0, 1);
            __builtin_prefetch(W + (size_t)(col0 + ldRow) * DIM + (kt + 2) * 32 + ldCol, 0, 1);
        }

        FragAB afrag[4], bfrag[2];
#pragma unroll
        for (int mt = 0; mt < 4; ++mt) {
            const bf16* p = &As[buf][(mbase + 16 * mt + l) * LDT + hi * 8];
            afrag[mt].h[0] = *(const bf16x8*)p;          // K = hi*8 .. +7
            afrag[mt].h[1] = *(const bf16x8*)(p + 16);   // K = 16+hi*8 .. +7
        }
#pragma unroll
        for (int nt = 0; nt < 2; ++nt) {
            const bf16* p = &Bs[buf][(nbase + 16 * nt + l) * LDT + hi * 8];
            bfrag[nt].h[0] = *(const bf16x8*)p;
            bfrag[nt].h[1] = *(const bf16x8*)(p + 16);
        }
#pragma unroll
        for (int mt = 0; mt < 4; ++mt)
#pragma unroll
            for (int nt = 0; nt < 2; ++nt)
                acc[mt][nt] = __builtin_amdgcn_wmma_f32_16x16x32_bf16(
                    false, afrag[mt].v, false, bfrag[nt].v,
                    (short)0, acc[mt][nt], false, false);

        if (kt + 1 < NK) sstore(buf ^ 1);
        __syncthreads();
    }

    // Store D.  C/D layout: VGPR r, lanes 0-15 -> M=r, lanes 16-31 -> M=8+r; N=lane%16.
    bf16* outp = (wsel == 0) ? qb : (wsel == 1) ? kb : vtb;
#pragma unroll
    for (int mt = 0; mt < 4; ++mt)
#pragma unroll
        for (int nt = 0; nt < 2; ++nt) {
            const int col = col0 + nbase + 16 * nt + l;
#pragma unroll
            for (int r = 0; r < 8; ++r) {
                const int row = row0 + mbase + 16 * mt + r + 8 * hi;   // batch index
                const float d = acc[mt][nt][r];
                if (wsel == 2) {
                    // v transposed: vT[b, f, h]  with n = h*64 + f
                    outp[(size_t)row * DIM + ((col & 63) << 6) + (col >> 6)] = (bf16)d;
                } else {
                    outp[(size_t)row * DIM + col] = (bf16)d;
                }
            }
        }
}

// ---------------------------------------------------------------------------
// Kernel 2: per-batch attention.  One block (128 thr = 4 waves) per batch.
//   S = (q k^T)/8 * mask ; P = softmax(S) ; wv = P v ; out = x * wv^T
// q/k/vT staging uses async global->LDS copies (ASYNCcnt path).
// ---------------------------------------------------------------------------
__launch_bounds__(128)
__global__ void attention_wmma(const float* __restrict__ x,
                               const float* __restrict__ imap,
                               const bf16* __restrict__ qb,
                               const bf16* __restrict__ kb,
                               const bf16* __restrict__ vtb,
                               float* __restrict__ out0,
                               float* __restrict__ aw)
{
    __shared__ bf16  qs [64 * 72];
    __shared__ bf16  ksh[64 * 72];
    __shared__ bf16  vsh[64 * 72];   // holds vT: vsh[f][g]
    __shared__ float Ss [64 * 65];
    __shared__ bf16  Ps [64 * 72];

    const int b   = blockIdx.x;
    const int tid = threadIdx.x;

    // cooperative async load of 3 x (64x64 bf16) tiles into padded LDS.
    // Per lane: one 16B global->LDS copy per tile, tracked by ASYNCcnt.
    {
        const int row  = tid >> 1;
        const int half = (tid & 1) * 32;
#pragma unroll
        for (int j = 0; j < 4; ++j) {
            const int c = half + 8 * j;
            const size_t g = (size_t)b * DIM + row * 64 + c;
            // generic pointers to __shared__ carry the LDS byte offset in
            // their low 32 bits (aperture in the high half)
            unsigned qoff = (unsigned)(uintptr_t)&qs [row * 72 + c];
            unsigned koff = (unsigned)(uintptr_t)&ksh[row * 72 + c];
            unsigned voff = (unsigned)(uintptr_t)&vsh[row * 72 + c];
            asm volatile("global_load_async_to_lds_b128 %0, %1, off"
                         :: "v"(qoff), "v"(qb + g) : "memory");
            asm volatile("global_load_async_to_lds_b128 %0, %1, off"
                         :: "v"(koff), "v"(kb + g) : "memory");
            asm volatile("global_load_async_to_lds_b128 %0, %1, off"
                         :: "v"(voff), "v"(vtb + g) : "memory");
        }
        asm volatile("s_wait_asynccnt 0x0" ::: "memory");
    }
    __syncthreads();

    const int lane = tid & 31;
    const int w    = tid >> 5;      // 4 waves -> 16 rows of S each
    const int h0   = w * 16;
    const int l    = lane & 15;
    const int hi   = lane >> 4;

    // ---- S = q k^T ----
    f32x8 accS[4] = {};
#pragma unroll
    for (int ks = 0; ks < 2; ++ks) {
        const int kb0 = ks * 32;
        FragAB aq;
        const bf16* ap = &qs[(h0 + l) * 72 + kb0 + hi * 8];
        aq.h[0] = *(const bf16x8*)ap;
        aq.h[1] = *(const bf16x8*)(ap + 16);
#pragma unroll
        for (int nt = 0; nt < 4; ++nt) {
            FragAB bk;
            const bf16* bp = &ksh[(16 * nt + l) * 72 + kb0 + hi * 8];
            bk.h[0] = *(const bf16x8*)bp;
            bk.h[1] = *(const bf16x8*)(bp + 16);
            accS[nt] = __builtin_amdgcn_wmma_f32_16x16x32_bf16(
                false, aq.v, false, bk.v, (short)0, accS[nt], false, false);
        }
    }
    // scale + multiplicative mask -> LDS (f32)
#pragma unroll
    for (int nt = 0; nt < 4; ++nt) {
        const int col = 16 * nt + l;
#pragma unroll
        for (int r = 0; r < 8; ++r) {
            const int rr = h0 + r + 8 * hi;
            Ss[rr * 65 + col] =
                accS[nt][r] * 0.125f * imap[(size_t)b * DIM + rr * 64 + col];
        }
    }
    __syncthreads();

    // ---- softmax rows (f32), emit attention_weights + bf16 P ----
    if (tid < 64) {
        float m = -3.0e38f;
        for (int c = 0; c < 64; ++c) m = fmaxf(m, Ss[tid * 65 + c]);
        float s = 0.f;
        for (int c = 0; c < 64; ++c) s += __expf(Ss[tid * 65 + c] - m);
        const float inv = 1.f / s;
        for (int c = 0; c < 64; ++c) {
            const float pv = __expf(Ss[tid * 65 + c] - m) * inv;
            Ps[tid * 72 + c] = (bf16)pv;
            aw[(size_t)b * DIM + tid * 64 + c] = pv;
        }
    }
    __syncthreads();

    // ---- wv = P v   (B operand = vT rows, contraction-contiguous) ----
    f32x8 accO[4] = {};
#pragma unroll
    for (int ks = 0; ks < 2; ++ks) {
        const int kb0 = ks * 32;
        FragAB apf;
        const bf16* pp = &Ps[(h0 + l) * 72 + kb0 + hi * 8];
        apf.h[0] = *(const bf16x8*)pp;
        apf.h[1] = *(const bf16x8*)(pp + 16);
#pragma unroll
        for (int nt = 0; nt < 4; ++nt) {
            FragAB bv;
            const bf16* bp = &vsh[(16 * nt + l) * 72 + kb0 + hi * 8];
            bv.h[0] = *(const bf16x8*)bp;
            bv.h[1] = *(const bf16x8*)(bp + 16);
            accO[nt] = __builtin_amdgcn_wmma_f32_16x16x32_bf16(
                false, apf.v, false, bv.v, (short)0, accO[nt], false, false);
        }
    }
    // out[b,0,i,j] = x[b,0,i,j] * wv[h=j, f=i]  (fused transpose + multiply)
#pragma unroll
    for (int nt = 0; nt < 4; ++nt) {
        const int f = 16 * nt + l;
#pragma unroll
        for (int r = 0; r < 8; ++r) {
            const int h   = h0 + r + 8 * hi;
            const size_t idx = (size_t)b * DIM + f * 64 + h;
            out0[idx] = x[idx] * accO[nt][r];
        }
    }
}

// ---------------------------------------------------------------------------
extern "C" void kernel_launch(void* const* d_in, const int* in_sizes, int n_in,
                              void* d_out, int out_size, void* d_ws, size_t ws_size,
                              hipStream_t stream) {
    const float* x  = (const float*)d_in[0];
    const float* im = (const float*)d_in[1];
    const float* Wq = (const float*)d_in[2];
    const float* Wk = (const float*)d_in[3];
    const float* Wv = (const float*)d_in[4];

    float* out0 = (float*)d_out;                   // output (512,1,64,64)
    float* aw   = out0 + (size_t)BATCH * DIM;      // attention_weights (512,64,64)

    bf16* qb  = (bf16*)d_ws;                       // (B, D) bf16   4 MB
    bf16* kbp = qb  + (size_t)BATCH * DIM;         // (B, D) bf16   4 MB
    bf16* vtb = kbp + (size_t)BATCH * DIM;         // (B, F, H) bf16 4 MB

    dim3 g1(BATCH / 128, 96);                      // 96 = 3 weights * 32 N-blocks
    qkv_gemm_wmma<<<g1, dim3(256), 0, stream>>>(x, Wq, Wk, Wv, qb, kbp, vtb);
    attention_wmma<<<dim3(BATCH), dim3(128), 0, stream>>>(x, im, qb, kbp, vtb, out0, aw);
}
